// Attention_63239098466657
// MI455X (gfx1250) — matrix operand order repaired and checked by reading, not compiled, not run on hardware
//
#include <hip/hip_runtime.h>
#include <hip/hip_bf16.h>

// Problem dimensions (fixed by the reference setup_inputs()).
constexpr int BATCH = 32;
constexpr int SEQ_T = 1024;
constexpr int SEQ_S = 1024;
constexpr int HID   = 1024;
constexpr int EMB   = 512;
constexpr float SCALE = 0.70710678118654752f; // sqrt(0.5)

typedef __bf16 bf16;
typedef __attribute__((ext_vector_type(16))) __bf16 v16bf;
typedef __attribute__((ext_vector_type(8)))  __bf16 v8bf;
typedef __attribute__((ext_vector_type(8)))  float  v8f;

__device__ __forceinline__ bf16 to_bf16(float x) {
    unsigned u = __builtin_bit_cast(unsigned, x);
    unsigned r = (u + 0x7FFFu + ((u >> 16) & 1u)) >> 16;   // round-to-nearest-even
    unsigned short h = (unsigned short)r;
    return __builtin_bit_cast(bf16, h);
}

// ---------------------------------------------------------------------------
// WMMA fragment loaders (wave32, v_wmma_f32_16x16x32_bf16)
//
// A matrix 16x32 (MxK) per-lane layout (ISA 7.12.2):
//   lanes 0-15 : M=lane,    K = 0..7  and 16..23
//   lanes 16-31: M=lane-16, K = 8..15 and 24..31
// B matrix 32x16 (KxN), stored here as N-major [N][K] (B^T row-major):
//   lanes 0-15 : N=lane,    K = 0..7  and 16..23
//   lanes 16-31: N=lane-16, K = 8..15 and 24..31
// ---------------------------------------------------------------------------
__device__ __forceinline__ v16bf load_frag_a(const bf16* __restrict__ base,
                                             int ld, int m0, int k0) {
    int lane = threadIdx.x & 31;
    const bf16* q = base + (size_t)(m0 + (lane & 15)) * ld + (k0 + ((lane >> 4) << 3));
    v8bf lo = *reinterpret_cast<const v8bf*>(q);
    v8bf hi = *reinterpret_cast<const v8bf*>(q + 16);
    v16bf f;
#pragma unroll
    for (int i = 0; i < 8; ++i) { f[i] = lo[i]; f[i + 8] = hi[i]; }
    return f;
}

__device__ __forceinline__ v16bf load_frag_b(const bf16* __restrict__ base,
                                             int ld, int n0, int k0) {
    int lane = threadIdx.x & 31;
    const bf16* q = base + (size_t)(n0 + (lane & 15)) * ld + (k0 + ((lane >> 4) << 3));
    v8bf lo = *reinterpret_cast<const v8bf*>(q);
    v8bf hi = *reinterpret_cast<const v8bf*>(q + 16);
    v16bf f;
#pragma unroll
    for (int i = 0; i < 8; ++i) { f[i] = lo[i]; f[i + 8] = hi[i]; }
    return f;
}

// ---------------------------------------------------------------------------
// Register-blocked macro-tile: one wave computes 32x64 of C
// (2 M-fragments x 4 N-fragments). Per K-step: 6 fragment loads -> 8 wmma.
// ---------------------------------------------------------------------------
__device__ __forceinline__ void gemm_tile_2x4(const bf16* __restrict__ A, int lda,
                                              const bf16* __restrict__ Bt, int ldb,
                                              int m0, int n0, int K,
                                              v8f acc[2][4]) {
#pragma unroll
    for (int i = 0; i < 2; ++i)
#pragma unroll
        for (int j = 0; j < 4; ++j)
            acc[i][j] = (v8f){0.f, 0.f, 0.f, 0.f, 0.f, 0.f, 0.f, 0.f};

#pragma unroll 2
    for (int k = 0; k < K; k += 32) {
        v16bf a0 = load_frag_a(A, lda, m0,      k);
        v16bf a1 = load_frag_a(A, lda, m0 + 16, k);
        v16bf b0 = load_frag_b(Bt, ldb, n0,      k);
        v16bf b1 = load_frag_b(Bt, ldb, n0 + 16, k);
        v16bf b2 = load_frag_b(Bt, ldb, n0 + 32, k);
        v16bf b3 = load_frag_b(Bt, ldb, n0 + 48, k);
        acc[0][0] = __builtin_amdgcn_wmma_f32_16x16x32_bf16(false, a0, false, b0, (short)0, acc[0][0], false, false);
        acc[0][1] = __builtin_amdgcn_wmma_f32_16x16x32_bf16(false, a0, false, b1, (short)0, acc[0][1], false, false);
        acc[0][2] = __builtin_amdgcn_wmma_f32_16x16x32_bf16(false, a0, false, b2, (short)0, acc[0][2], false, false);
        acc[0][3] = __builtin_amdgcn_wmma_f32_16x16x32_bf16(false, a0, false, b3, (short)0, acc[0][3], false, false);
        acc[1][0] = __builtin_amdgcn_wmma_f32_16x16x32_bf16(false, a1, false, b0, (short)0, acc[1][0], false, false);
        acc[1][1] = __builtin_amdgcn_wmma_f32_16x16x32_bf16(false, a1, false, b1, (short)0, acc[1][1], false, false);
        acc[1][2] = __builtin_amdgcn_wmma_f32_16x16x32_bf16(false, a1, false, b2, (short)0, acc[1][2], false, false);
        acc[1][3] = __builtin_amdgcn_wmma_f32_16x16x32_bf16(false, a1, false, b3, (short)0, acc[1][3], false, false);
    }
}

// Wave->macro-tile mapping inside a 256-thread block (8 waves as 2x4):
// block tile = 64 (M) x 256 (N).
__device__ __forceinline__ void tile_origin(int& m0, int& n0) {
    int wave = threadIdx.x >> 5;
    m0 = blockIdx.x * 64 + (wave >> 2) * 32;
    n0 = blockIdx.y * 256 + (wave & 3) * 64;
}

// ---------------------------------------------------------------------------
// Prep kernels: cast f32->bf16, and fused transpose+cast.
// ---------------------------------------------------------------------------
__global__ void k_cast_bf16(const float* __restrict__ src, bf16* __restrict__ dst,
                            size_t n) {
    size_t i = (size_t)blockIdx.x * blockDim.x + threadIdx.x;
    if (i < n) dst[i] = to_bf16(src[i]);
}

// src: per-batch [R,C] f32 -> dst: per-batch [C,R] bf16
__global__ void k_transpose_cast(const float* __restrict__ src, bf16* __restrict__ dst,
                                 int R, int C) {
    __shared__ float tile[32][33];
    int b = blockIdx.z;
    const float* s = src + (size_t)b * R * C;
    bf16* d = dst + (size_t)b * R * C;
    int c0 = blockIdx.x * 32, r0 = blockIdx.y * 32;
#pragma unroll
    for (int i = threadIdx.y; i < 32; i += 8)
        tile[i][threadIdx.x] = s[(size_t)(r0 + i) * C + (c0 + threadIdx.x)];
    __syncthreads();
#pragma unroll
    for (int i = threadIdx.y; i < 32; i += 8)
        d[(size_t)(c0 + i) * R + (r0 + threadIdx.x)] = to_bf16(tile[threadIdx.x][i]);
}

// ---------------------------------------------------------------------------
// GEMM 1: combined[b,t,e] = (convedT[b,t,:]·W_h2e[e,:] + b_h2e[e] + embedded)·SCALE
// ---------------------------------------------------------------------------
__global__ void __launch_bounds__(256, 1)
k_proj_combine(const bf16* __restrict__ convedT,
               const bf16* __restrict__ Wh2e,
               const float* __restrict__ bias,
               const float* __restrict__ embedded,
               bf16* __restrict__ combined) {
    int lane = threadIdx.x & 31;
    int b = blockIdx.z;
    int m0, n0; tile_origin(m0, n0);
    const bf16* A = convedT + (size_t)b * SEQ_T * HID;
    v8f acc[2][4];
    gemm_tile_2x4(A, HID, Wh2e, HID, m0, n0, HID, acc);
#pragma unroll
    for (int i = 0; i < 2; ++i) {
        int mb = m0 + i * 16 + ((lane >> 4) << 3);
#pragma unroll
        for (int j = 0; j < 4; ++j) {
            int n = n0 + j * 16 + (lane & 15);
            float bn = bias[n];
#pragma unroll
            for (int r = 0; r < 8; ++r) {
                size_t idx = ((size_t)b * SEQ_T + (mb + r)) * EMB + n;
                combined[idx] = to_bf16((acc[i][j][r] + bn + embedded[idx]) * SCALE);
            }
        }
    }
}

// ---------------------------------------------------------------------------
// GEMM 2: energy[b,t,s] = combined[t,:]·enc_conved[s,:] -> f32 into d_out
// ---------------------------------------------------------------------------
__global__ void __launch_bounds__(256, 1)
k_energy(const bf16* __restrict__ combined,
         const bf16* __restrict__ encconv,
         float* __restrict__ energy) {
    int lane = threadIdx.x & 31;
    int b = blockIdx.z;
    int m0, n0; tile_origin(m0, n0);
    const bf16* A  = combined + (size_t)b * SEQ_T * EMB;
    const bf16* Bt = encconv  + (size_t)b * SEQ_S * EMB;
    v8f acc[2][4];
    gemm_tile_2x4(A, EMB, Bt, EMB, m0, n0, EMB, acc);
#pragma unroll
    for (int i = 0; i < 2; ++i) {
        int mb = m0 + i * 16 + ((lane >> 4) << 3);
#pragma unroll
        for (int j = 0; j < 4; ++j) {
            int n = n0 + j * 16 + (lane & 15);
#pragma unroll
            for (int r = 0; r < 8; ++r)
                energy[((size_t)b * SEQ_T + (mb + r)) * SEQ_S + n] = acc[i][j][r];
        }
    }
}

// ---------------------------------------------------------------------------
// Softmax over S=1024, in place on d_out; also emit bf16 copy for GEMM 3.
// ---------------------------------------------------------------------------
__global__ void k_softmax(float* __restrict__ att, bf16* __restrict__ att_bf) {
    __shared__ float red[256];
    size_t row = blockIdx.x;
    float* p = att + row * (size_t)SEQ_S;
    int tid = threadIdx.x;
    float v[4];
    float lmax = -3.402823466e38f;
#pragma unroll
    for (int i = 0; i < 4; ++i) { v[i] = p[tid + 256 * i]; lmax = fmaxf(lmax, v[i]); }
    red[tid] = lmax; __syncthreads();
    for (int s = 128; s > 0; s >>= 1) {
        if (tid < s) red[tid] = fmaxf(red[tid], red[tid + s]);
        __syncthreads();
    }
    float m = red[0];
    __syncthreads();
    float lsum = 0.f;
#pragma unroll
    for (int i = 0; i < 4; ++i) { v[i] = expf(v[i] - m); lsum += v[i]; }
    red[tid] = lsum; __syncthreads();
    for (int s = 128; s > 0; s >>= 1) {
        if (tid < s) red[tid] += red[tid + s];
        __syncthreads();
    }
    float inv = 1.0f / red[0];
    bf16* q = att_bf + row * (size_t)SEQ_S;
#pragma unroll
    for (int i = 0; i < 4; ++i) {
        float a = v[i] * inv;
        p[tid + 256 * i] = a;
        q[tid + 256 * i] = to_bf16(a);
    }
}

// ---------------------------------------------------------------------------
// GEMM 3: attn_enc[b,t,e] = attention[t,:]·enc_combined[:,e]   (bf16 out)
// ---------------------------------------------------------------------------
__global__ void __launch_bounds__(256, 1)
k_context(const bf16* __restrict__ attn,
          const bf16* __restrict__ enccombT,
          bf16* __restrict__ attnenc) {
    int lane = threadIdx.x & 31;
    int b = blockIdx.z;
    int m0, n0; tile_origin(m0, n0);
    const bf16* A  = attn     + (size_t)b * SEQ_T * SEQ_S;
    const bf16* Bt = enccombT + (size_t)b * EMB * SEQ_S;
    v8f acc[2][4];
    gemm_tile_2x4(A, SEQ_S, Bt, SEQ_S, m0, n0, SEQ_S, acc);
#pragma unroll
    for (int i = 0; i < 2; ++i) {
        int mb = m0 + i * 16 + ((lane >> 4) << 3);
#pragma unroll
        for (int j = 0; j < 4; ++j) {
            int n = n0 + j * 16 + (lane & 15);
#pragma unroll
            for (int r = 0; r < 8; ++r)
                attnenc[((size_t)b * SEQ_T + (mb + r)) * EMB + n] = to_bf16(acc[i][j][r]);
        }
    }
}

// ---------------------------------------------------------------------------
// GEMM 4: out2[b,h,t] = (W_e2h[h,:]·attn_enc[t,:] + b_e2h[h] + conved[b,h,t])·SCALE
// ---------------------------------------------------------------------------
__global__ void __launch_bounds__(256, 1)
k_outproj(const bf16* __restrict__ We2h,
          const bf16* __restrict__ attnenc,
          const float* __restrict__ bias,
          const float* __restrict__ conved,
          float* __restrict__ out2) {
    int lane = threadIdx.x & 31;
    int b = blockIdx.z;
    int m0, n0; tile_origin(m0, n0);
    const bf16* Bt = attnenc + (size_t)b * SEQ_T * EMB;
    v8f acc[2][4];
    gemm_tile_2x4(We2h, EMB, Bt, EMB, m0, n0, EMB, acc);
#pragma unroll
    for (int i = 0; i < 2; ++i) {
        int mb = m0 + i * 16 + ((lane >> 4) << 3);
#pragma unroll
        for (int j = 0; j < 4; ++j) {
            int n = n0 + j * 16 + (lane & 15);
#pragma unroll
            for (int r = 0; r < 8; ++r) {
                int m = mb + r;
                size_t idx = ((size_t)b * HID + m) * SEQ_T + n;
                out2[idx] = (acc[i][j][r] + bias[m] + conved[idx]) * SCALE;
            }
        }
    }
}

// ---------------------------------------------------------------------------
// Launcher
// ---------------------------------------------------------------------------
extern "C" void kernel_launch(void* const* d_in, const int* in_sizes, int n_in,
                              void* d_out, int out_size, void* d_ws, size_t ws_size,
                              hipStream_t stream) {
    (void)in_sizes; (void)n_in; (void)out_size; (void)ws_size;

    const float* embedded     = (const float*)d_in[0];   // [B,T,E]
    const float* conved       = (const float*)d_in[1];   // [B,H,T]
    const float* enc_conved   = (const float*)d_in[2];   // [B,S,E]
    const float* enc_combined = (const float*)d_in[3];   // [B,S,E]
    const float* W_h2e        = (const float*)d_in[4];   // [E,H]
    const float* b_h2e        = (const float*)d_in[5];   // [E]
    const float* W_e2h        = (const float*)d_in[6];   // [H,E]
    const float* b_e2h        = (const float*)d_in[7];   // [H]

    float* out_att  = (float*)d_out;                              // [B,T,S]
    float* out_comb = out_att + (size_t)BATCH * SEQ_T * SEQ_S;    // [B,H,T]

    // Workspace layout (bf16 staging), 256B-aligned chunks.
    char* w = (char*)d_ws;
    auto take = [&](size_t bytes) {
        char* p = w;
        w += (bytes + 255) & ~(size_t)255;
        return p;
    };
    bf16* convedT_bf = (bf16*)take((size_t)BATCH * SEQ_T * HID * 2);   // [B,T,H]
    bf16* encconv_bf = (bf16*)take((size_t)BATCH * SEQ_S * EMB * 2);   // [B,S,E]
    bf16* enccombT_bf= (bf16*)take((size_t)BATCH * EMB * SEQ_S * 2);   // [B,E,S]
    bf16* combined_bf= (bf16*)take((size_t)BATCH * SEQ_T * EMB * 2);   // [B,T,E]
    bf16* attn_bf    = (bf16*)take((size_t)BATCH * SEQ_T * SEQ_S * 2); // [B,T,S]
    bf16* attnenc_bf = (bf16*)take((size_t)BATCH * SEQ_T * EMB * 2);   // [B,T,E]
    bf16* Wh2e_bf    = (bf16*)take((size_t)EMB * HID * 2);             // [E,H]
    bf16* We2h_bf    = (bf16*)take((size_t)HID * EMB * 2);             // [H,E]

    dim3 tp(32, 8);

    // Prep: casts + transposes
    k_transpose_cast<<<dim3(SEQ_T / 32, HID / 32, BATCH), tp, 0, stream>>>(
        conved, convedT_bf, HID, SEQ_T);                 // [H,T] -> [T,H]
    {
        size_t n = (size_t)BATCH * SEQ_S * EMB;
        k_cast_bf16<<<(unsigned)((n + 255) / 256), 256, 0, stream>>>(enc_conved, encconv_bf, n);
    }
    k_transpose_cast<<<dim3(EMB / 32, SEQ_S / 32, BATCH), tp, 0, stream>>>(
        enc_combined, enccombT_bf, SEQ_S, EMB);          // [S,E] -> [E,S]
    {
        size_t n = (size_t)EMB * HID;
        k_cast_bf16<<<(unsigned)((n + 255) / 256), 256, 0, stream>>>(W_h2e, Wh2e_bf, n);
        k_cast_bf16<<<(unsigned)((n + 255) / 256), 256, 0, stream>>>(W_e2h, We2h_bf, n);
    }

    // GEMM 1: M=T, N=E, K=H
    k_proj_combine<<<dim3(SEQ_T / 64, EMB / 256, BATCH), 256, 0, stream>>>(
        convedT_bf, Wh2e_bf, b_h2e, embedded, combined_bf);

    // GEMM 2: M=T, N=S, K=E  (f32 energy straight into d_out)
    k_energy<<<dim3(SEQ_T / 64, SEQ_S / 256, BATCH), 256, 0, stream>>>(
        combined_bf, encconv_bf, out_att);

    // Softmax in place + bf16 copy
    k_softmax<<<BATCH * SEQ_T, 256, 0, stream>>>(out_att, attn_bf);

    // GEMM 3: M=T, N=E, K=S
    k_context<<<dim3(SEQ_T / 64, EMB / 256, BATCH), 256, 0, stream>>>(
        attn_bf, enccombT_bf, attnenc_bf);

    // GEMM 4: M=H, N=T, K=E
    k_outproj<<<dim3(HID / 64, SEQ_T / 256, BATCH), 256, 0, stream>>>(
        We2h_bf, attnenc_bf, b_e2h, conved, out_comb);
}
